// TransformerAttention_3702261809284
// MI455X (gfx1250) — matrix operand order repaired and checked by reading, not compiled, not run on hardware
//
#include <hip/hip_runtime.h>
#include <hip/hip_bf16.h>

typedef __attribute__((ext_vector_type(16))) __bf16 v16bf;
typedef __attribute__((ext_vector_type(8)))  __bf16 v8bf;
typedef __attribute__((ext_vector_type(4)))  __bf16 v4bf;
typedef __attribute__((ext_vector_type(8)))  float  v8f;
typedef __attribute__((ext_vector_type(4)))  float  v4f;
typedef __attribute__((ext_vector_type(4)))  unsigned int v4u;
typedef __attribute__((ext_vector_type(8)))  int    v8i;
typedef __attribute__((ext_vector_type(4)))  int    v4i;

#if defined(__has_builtin)
#if __has_builtin(__builtin_amdgcn_tensor_load_to_lds) && \
    __has_builtin(__builtin_amdgcn_s_wait_tensorcnt)
#define USE_TDM 1
#endif
#endif

__device__ __forceinline__ v16bf cat8(v8bf lo, v8bf hi) {
    return __builtin_shufflevector(lo, hi, 0,1,2,3,4,5,6,7,8,9,10,11,12,13,14,15);
}

// ---------------------------------------------------------------------------
// C[M,N] = A[M,K] * W[N,K]^T  (einsum '...i,oi->...o'), bf16 WMMA, fp32 accum.
// Block tile 128x128, K-step 32.  8 waves: 2(m) x 4(n), each wave 4x2 WMMA
// tiles (64x32 output) -> 8 v_wmma per k-step per wave.
// MODE 0: fp32 store to Cf[M,N]
// MODE 1: fused RoPE -> bf16 store to Cb[B,H,T,Hd]
// MODE 2: bf16 store to Cb[B,H,T,Hd] (no RoPE)
// ---------------------------------------------------------------------------
#define GBM 128
#define GBN 128
#define GBK 32
#define GLD 40          // 32 + 8 pad -> 80B rows, keeps 16B-aligned b128 chunks

template<int MODE>
__global__ __launch_bounds__(256)
void gemm_xwt(const float* __restrict__ A, const float* __restrict__ W,
              float* __restrict__ Cf, __bf16* __restrict__ Cb,
              int M, int N, int K, int T, int Hd)
{
    __shared__ __bf16 As[GBM][GLD];
    __shared__ __bf16 Ws[GBN][GLD];

    const int tid  = threadIdx.x;
    const int lane = tid & 31;
    const int wave = tid >> 5;      // 0..7
    const int wm   = wave >> 2;     // 0..1 -> 64-row slab
    const int wn   = wave & 3;      // 0..3 -> 32-col slab
    const int g    = lane >> 4;     // half-wave
    const int ln   = lane & 15;
    const int m0   = blockIdx.y * GBM;
    const int n0   = blockIdx.x * GBN;

    v8f acc[4][2] = {};

    for (int k0 = 0; k0 < K; k0 += GBK) {
        // stage A tile 128x32 (1024 float4, 4/thread), convert fp32 -> bf16
        #pragma unroll
        for (int i = 0; i < 4; ++i) {
            int idx = tid + i * 256;
            int r = idx >> 3, c = (idx & 7) * 4;
            v4f f = *(const v4f*)&A[(size_t)(m0 + r) * K + k0 + c];
            v4bf h;
            #pragma unroll
            for (int e = 0; e < 4; ++e) h[e] = (__bf16)f[e];
            *(v4bf*)&As[r][c] = h;
        }
        // stage W tile 128x32
        #pragma unroll
        for (int i = 0; i < 4; ++i) {
            int idx = tid + i * 256;
            int r = idx >> 3, c = (idx & 7) * 4;
            v4f f = *(const v4f*)&W[(size_t)(n0 + r) * K + k0 + c];
            v4bf h;
            #pragma unroll
            for (int e = 0; e < 4; ++e) h[e] = (__bf16)f[e];
            *(v4bf*)&Ws[r][c] = h;
        }
        // prefetch next k-tile (global_prefetch_b8)
        if (k0 + GBK < K) {
            __builtin_prefetch(&A[(size_t)(m0 + (tid >> 1)) * K + k0 + GBK + (tid & 1) * 16], 0, 0);
            __builtin_prefetch(&W[(size_t)(n0 + (tid >> 1)) * K + k0 + GBK + (tid & 1) * 16], 0, 0);
        }
        __syncthreads();

        v16bf a[4], b[2];
        #pragma unroll
        for (int i = 0; i < 4; ++i) {          // A frag: row=lane&15, chunks g*8 / 16+g*8
            int r = wm * 64 + i * 16 + ln;
            v8bf lo = *(const v8bf*)&As[r][g * 8];
            v8bf hi = *(const v8bf*)&As[r][16 + g * 8];
            a[i] = cat8(lo, hi);
        }
        #pragma unroll
        for (int j = 0; j < 2; ++j) {          // B frag: col=lane&15, chunks g*8 / 16+g*8
            int r = wn * 32 + j * 16 + ln;
            v8bf lo = *(const v8bf*)&Ws[r][g * 8];
            v8bf hi = *(const v8bf*)&Ws[r][16 + g * 8];
            b[j] = cat8(lo, hi);
        }
        #pragma unroll
        for (int i = 0; i < 4; ++i)
            #pragma unroll
            for (int j = 0; j < 2; ++j)
                acc[i][j] = __builtin_amdgcn_wmma_f32_16x16x32_bf16(
                    false, a[i], false, b[j], (short)0, acc[i][j], false, false);
        __syncthreads();
    }

    const int H = N / Hd;
    #pragma unroll
    for (int j = 0; j < 2; ++j) {
        const int col = n0 + wn * 32 + j * 16 + ln;
        const int d   = col & (Hd - 1);
        const int h   = col / Hd;
        float inv = 0.0f;
        if (MODE == 1)
            inv = __powf(10000.0f, -((float)(2 * (d >> 1)) / (float)Hd));
        #pragma unroll
        for (int i = 0; i < 4; ++i) {
            const int rowb = m0 + wm * 64 + i * 16 + g * 8;
            #pragma unroll
            for (int r = 0; r < 8; ++r) {
                const int row = rowb + r;
                float val = acc[i][j][r];
                if (MODE == 0) {
                    Cf[(size_t)row * N + col] = val;
                } else {
                    const int t    = row % T;
                    const int bidx = row / T;
                    float o = val;
                    if (MODE == 1) {
                        // partner feature (d^1) lives in lane ln^1, same row/tile
                        float other = __shfl_xor(val, 1, 32);
                        float s, c;
                        __sincosf((float)t * inv, &s, &c);
                        // even d: x1*c - x2*s ; odd d: x1*s + x2*c (x1=even,x2=odd)
                        o = val * c + ((d & 1) ? other * s : -other * s);
                    }
                    Cb[(((size_t)(bidx * H + h)) * T + t) * Hd + d] = (__bf16)o;
                }
            }
        }
    }
}

// ---------------------------------------------------------------------------
// Causal flash attention: block = (b, h, 64 q-rows), 128 threads = 4 waves,
// wave w owns q rows q0+w*16..+15.  kv processed 32 at a time.
// K tile staged by the Tensor Data Mover (TDM) when available.
// ---------------------------------------------------------------------------
#define AQB 64
#define AKB 32
#define AHD 64
#define ALK 72          // 64 + 8 pad (bf16) -> 144B rows; matches TDM pad 4dw/32dw
#define ALV 40
#define ALP 40

__global__ __launch_bounds__(128)
void attn(const __bf16* __restrict__ qb, const __bf16* __restrict__ kb,
          const __bf16* __restrict__ vb, float* __restrict__ xo,
          int Bn, int H, int T, float scale)
{
    __shared__ __bf16 Ks[AKB][ALK];        // [kv][hd]
    __shared__ __bf16 Vt[AHD][ALV];        // [hd][kv] (transposed)
    __shared__ __bf16 Ps[4][16][ALP];      // per-wave P staging

    const int tid  = threadIdx.x;
    const int lane = tid & 31;
    const int wave = tid >> 5;             // 0..3
    const int g    = lane >> 4;
    const int ln   = lane & 15;
    const int q0   = blockIdx.x * AQB;
    const int h    = blockIdx.y;
    const int b    = blockIdx.z;

    const __bf16* Qg = qb + ((size_t)(b * H + h)) * T * AHD;
    const __bf16* Kg = kb + ((size_t)(b * H + h)) * T * AHD;
    const __bf16* Vg = vb + ((size_t)(b * H + h)) * T * AHD;

    // Q A-fragments for this wave (row = q0 + wave*16 + (lane&15))
    const int qrow = q0 + wave * 16 + ln;
    v16bf qa[2];
    #pragma unroll
    for (int cc = 0; cc < 2; ++cc) {
        v8bf lo = *(const v8bf*)&Qg[(size_t)qrow * AHD + cc * 32 + g * 8];
        v8bf hi = *(const v8bf*)&Qg[(size_t)qrow * AHD + cc * 32 + 16 + g * 8];
        qa[cc] = cat8(lo, hi);
    }

    v8f oacc[4] = {};
    float mi[8], li[8];
    #pragma unroll
    for (int r = 0; r < 8; ++r) { mi[r] = -3.0e38f; li[r] = 0.0f; }

    const int kv_end = q0 + AQB;           // causal: never need kv >= q0+64
    for (int kv0 = 0; kv0 < kv_end; kv0 += AKB) {
#ifdef USE_TDM
        // --- K tile via Tensor Data Mover: 32x64 bf16 2-D tile with LDS pad ---
        if (wave == 0) {
            unsigned lds_off = (unsigned)(size_t)&Ks[0][0];         // low 32 bits of
                                                                    // generic LDS ptr
            unsigned long long ga =
                (unsigned long long)(size_t)(Kg + (size_t)kv0 * AHD);
            v4u g0;
            g0[0] = 1u;                                             // count = 1
            g0[1] = lds_off;                                        // lds_addr
            g0[2] = (unsigned)ga;                                   // global_addr lo
            g0[3] = ((unsigned)(ga >> 32) & 0x1FFFFFFu) | (2u << 30); // addr hi | type=2
            v8i g1;
            g1[0] = (int)((1u << 16)        // data_size = 2B
                        | (1u << 20)        // pad_enable
                        | (4u << 22)        // pad_interval: 32 DWORDs (=64 bf16 row)
                        | (3u << 25));      // pad_amount: 4 DWORDs  (=8 bf16 pad)
            g1[1] = (int)(64u << 16);                               // tensor_dim0=64
            g1[2] = (int)(((unsigned)T & 0xFFFFu) << 16);           // tensor_dim1 lo
            g1[3] = (int)(((unsigned)T >> 16) | (64u << 16));       // dim1 hi | tile_dim0=64
            g1[4] = (int)32u;                                       // tile_dim1=32
            g1[5] = (int)64u;                                       // tensor_dim0_stride=64
            g1[6] = 0;
            g1[7] = 0;
            v4i gz4 = {0, 0, 0, 0};
            v8i gz8 = {0, 0, 0, 0, 0, 0, 0, 0};
            __builtin_amdgcn_tensor_load_to_lds(g0, g1, gz4, gz4, gz8, 0);
            __builtin_amdgcn_s_wait_tensorcnt(0);
        }
#else
        #pragma unroll
        for (int i = 0; i < 2; ++i) {
            int c = tid + i * 128;                  // 256 chunks of 8 bf16
            int r = c >> 3, cc = (c & 7) * 8;
            v8bf d = *(const v8bf*)&Kg[(size_t)(kv0 + r) * AHD + cc];
            *(v8bf*)&Ks[r][cc] = d;
        }
#endif
        // --- V tile transposed into LDS: [hd][kv] ---
        #pragma unroll
        for (int i = 0; i < 2; ++i) {
            int c = tid + i * 128;
            int r = c >> 3, cc = (c & 7) * 8;
            v8bf d = *(const v8bf*)&Vg[(size_t)(kv0 + r) * AHD + cc];
            #pragma unroll
            for (int e = 0; e < 8; ++e) Vt[cc + e][r] = d[e];
        }
        __syncthreads();

        const int qtop = q0 + wave * 16 + 15;
        if (kv0 <= qtop) {                          // tile not fully masked
            // --- scores S = Q K^T (Hd=64 contraction = 2 WMMAs per 16-col tile) ---
            v8f s[2] = {};
            #pragma unroll
            for (int j = 0; j < 2; ++j)
                #pragma unroll
                for (int cc = 0; cc < 2; ++cc) {
                    int r = j * 16 + ln;
                    v8bf lo = *(const v8bf*)&Ks[r][cc * 32 + g * 8];
                    v8bf hi = *(const v8bf*)&Ks[r][cc * 32 + 16 + g * 8];
                    v16bf bf = cat8(lo, hi);
                    s[j] = __builtin_amdgcn_wmma_f32_16x16x32_bf16(
                        false, qa[cc], false, bf, (short)0, s[j], false, false);
                }

            // --- scale + causal mask + per-row online softmax ---
            float p[2][8], rmax[8], rsum[8];
            #pragma unroll
            for (int r = 0; r < 8; ++r) {
                int qg = q0 + wave * 16 + g * 8 + r;
                float best = -3.0e38f;
                #pragma unroll
                for (int j = 0; j < 2; ++j) {
                    int kg = kv0 + j * 16 + ln;
                    float val = s[j][r] * scale;
                    if (kg > qg) val = -3.0e38f;
                    p[j][r] = val;
                    best = fmaxf(best, val);
                }
                rmax[r] = best;
            }
            #pragma unroll
            for (int off = 1; off < 16; off <<= 1)
                #pragma unroll
                for (int r = 0; r < 8; ++r)
                    rmax[r] = fmaxf(rmax[r], __shfl_xor(rmax[r], off, 16));

            float mnew[8], sc[8];
            #pragma unroll
            for (int r = 0; r < 8; ++r) {
                mnew[r] = fmaxf(mi[r], rmax[r]);
                sc[r]   = __expf(mi[r] - mnew[r]);
            }
            #pragma unroll
            for (int r = 0; r < 8; ++r) {
                float su = 0.0f;
                #pragma unroll
                for (int j = 0; j < 2; ++j) {
                    float e = __expf(p[j][r] - mnew[r]);
                    p[j][r] = e;
                    su += e;
                }
                rsum[r] = su;
            }
            #pragma unroll
            for (int off = 1; off < 16; off <<= 1)
                #pragma unroll
                for (int r = 0; r < 8; ++r)
                    rsum[r] += __shfl_xor(rsum[r], off, 16);
            #pragma unroll
            for (int r = 0; r < 8; ++r) {
                li[r] = li[r] * sc[r] + rsum[r];
                mi[r] = mnew[r];
            }
            #pragma unroll
            for (int t4 = 0; t4 < 4; ++t4)
                #pragma unroll
                for (int r = 0; r < 8; ++r)
                    oacc[t4][r] *= sc[r];

            // --- C-layout -> A-fragment relayout of P through per-wave LDS ---
            #pragma unroll
            for (int j = 0; j < 2; ++j)
                #pragma unroll
                for (int r = 0; r < 8; ++r)
                    Ps[wave][g * 8 + r][j * 16 + ln] = (__bf16)p[j][r];
            v8bf plo = *(const v8bf*)&Ps[wave][ln][g * 8];
            v8bf phi = *(const v8bf*)&Ps[wave][ln][16 + g * 8];
            v16bf pa = cat8(plo, phi);

            // --- O += P V  (kv=32 contraction, 4 hd tiles) ---
            #pragma unroll
            for (int t4 = 0; t4 < 4; ++t4) {
                int r = t4 * 16 + ln;                   // hd row in Vt
                v8bf lo = *(const v8bf*)&Vt[r][g * 8];
                v8bf hi = *(const v8bf*)&Vt[r][16 + g * 8];
                v16bf bv = cat8(lo, hi);
                oacc[t4] = __builtin_amdgcn_wmma_f32_16x16x32_bf16(
                    false, pa, false, bv, (short)0, oacc[t4], false, false);
            }
        }
        __syncthreads();
    }

    // epilogue: write normalized O into [B,T,H*Hd] fp32 for the final GEMM
    #pragma unroll
    for (int t4 = 0; t4 < 4; ++t4)
        #pragma unroll
        for (int r = 0; r < 8; ++r) {
            int qg  = q0 + wave * 16 + g * 8 + r;
            int col = h * AHD + t4 * 16 + ln;
            xo[((size_t)(b * T + qg)) * (H * AHD) + col] = oacc[t4][r] / li[r];
        }
}

// ---------------------------------------------------------------------------
extern "C" void kernel_launch(void* const* d_in, const int* in_sizes, int n_in,
                              void* d_out, int out_size, void* d_ws, size_t ws_size,
                              hipStream_t stream)
{
    const float* x  = (const float*)d_in[0];
    const float* wq = (const float*)d_in[1];
    const float* wk = (const float*)d_in[2];
    const float* wv = (const float*)d_in[3];
    const float* wo = (const float*)d_in[4];
    float* out = (float*)d_out;

    const int Bn = 2, T = 2048, Dm = 1024, H = 16, Hd = 64;
    const int M = Bn * T;                       // 4096

    char* ws = (char*)d_ws;
    size_t off = 0;
    float*  xof = (float*)(ws + off);  off += (size_t)M * Dm * sizeof(float);
    __bf16* qbp = (__bf16*)(ws + off); off += (size_t)M * Dm * sizeof(__bf16);
    __bf16* kbp = (__bf16*)(ws + off); off += (size_t)M * Dm * sizeof(__bf16);
    __bf16* vbp = (__bf16*)(ws + off); off += (size_t)M * Dm * sizeof(__bf16);

    dim3 gg(Dm / GBN, M / GBM);                 // (8, 32)
    // Q/K projections with fused RoPE + bf16 [B,H,T,Hd] repack
    gemm_xwt<1><<<gg, 256, 0, stream>>>(x, wq, nullptr, qbp, M, Dm, Dm, T, Hd);
    gemm_xwt<1><<<gg, 256, 0, stream>>>(x, wk, nullptr, kbp, M, Dm, Dm, T, Hd);
    // V projection: bf16 repack only
    gemm_xwt<2><<<gg, 256, 0, stream>>>(x, wv, nullptr, vbp, M, Dm, Dm, T, Hd);

    dim3 ga(T / AQB, H, Bn);                    // (32, 16, 2)
    attn<<<ga, 128, 0, stream>>>(qbp, kbp, vbp, xof, Bn, H, T, 0.125f);

    // output projection, fp32 straight into d_out
    gemm_xwt<0><<<gg, 256, 0, stream>>>(xof, wo, out, nullptr, M, Dm, Dm, T, Hd);
}
